// KarcherFFN_66254165508451
// MI455X (gfx1250) — matrix operand order repaired and checked by reading, not compiled, not run on hardware
//
#include <hip/hip_runtime.h>
#include <hip/hip_bf16.h>
#include <math.h>

// Problem constants (match reference)
#define BQ   2
#define SQ   1024
#define KQ   8
#define NB   32
#define DM   256
#define DH   1024
#define EPSF 1e-6f

typedef _Float16 f16_t;
typedef __attribute__((ext_vector_type(16))) _Float16 v16h;
typedef __attribute__((ext_vector_type(8)))  float    v8f;

// ---------------- Dynamic LDS layout (float offsets) ----------------
#define OFF_REC   0        // 256  : softmaxed recipes [8][32]
#define OFF_WK    256      // 8    : normalized neuron weights
#define OFF_NRM   264      // 8
#define OFF_DOT   272      // 8
#define OFF_COEF  280      // 8
#define OFF_RED   288      // 256  : reduction scratch
#define OFF_XF    544      // 256  : x tile [16][16]; reused for h [8][8]
#define OFF_CORE  800      // 6144 : A1@0 A2@1024 B1@2048 B2@4096
#define OFF_MU    6944     // 2048
#define OFF_VV    8992     // 2048
#define OFF_U     11040    // 16384: mixed/normalized cores u[8][D]; reused for t1/t2
#define LDS_FLOATS (11040 + 16384)   // 27424 floats = 109,696 bytes

__device__ __forceinline__ float block_reduce_sum(float val, float* red) {
  const int tid = threadIdx.x;
  red[tid] = val;
  __syncthreads();
  #pragma unroll
  for (int s = 128; s > 0; s >>= 1) {
    if (tid < s) red[tid] += red[tid + s];
    __syncthreads();
  }
  float r = red[0];
  __syncthreads();
  return r;
}

// Mixing GEMM via WMMA: u[k][d] = sum_n rec[k][n] * bas[n][d]  for k=0..7.
// afrag holds rec as a 16x32 f16 A-tile (rows 8..15 zero), shared by all tiles.
// Each wave owns d-tiles (D/16 tiles, uniform across the wave -> EXEC all-ones
// at every WMMA). C-tile rows m=0..7 live in the hi==0 half-lanes (VGPR j = k).
__device__ __forceinline__ void mix_wmma(const float* __restrict__ bas, int D,
                                         const v16h afrag, float* __restrict__ u) {
  const int tid  = threadIdx.x;
  const int lane = tid & 31;
  const int wave = tid >> 5;
  const int col  = lane & 15;
  const int hi   = lane >> 4;
  const int ntiles = D >> 4;          // 64 or 128; divisible by 8 waves
  for (int t = wave; t < ntiles; t += 8) {
    const int d0 = t * 16;
    v16h b;
    #pragma unroll
    for (int e = 0; e < 16; ++e) {
      const int n = (e < 8) ? (hi * 8 + e) : (16 + hi * 8 + (e - 8));  // B fragment: K-row in table order, N-col = col
      b[e] = (f16_t)bas[n * D + d0 + col];
    }
    v8f acc = {};
    acc = __builtin_amdgcn_wmma_f32_16x16x32_f16(false, afrag, false, b,
                                                 (short)0, acc, false, false);
    if (hi == 0) {                    // rows m = j = neuron slot k (0..7)
      #pragma unroll
      for (int j = 0; j < 8; ++j) u[j * D + d0 + col] = acc[j];
    }
  }
  __syncthreads();
}

// Weighted Karcher mean on the sphere for one TT-core tensor (D <= 2048).
__device__ void karcher_one(const float* __restrict__ bas, int D, float* coreOut,
                            const v16h afrag, const float* wk,
                            float* nrm, float* dotv, float* coef,
                            float* red, float* mu, float* vv, float* u) {
  const int tid = threadIdx.x;

  // 1) mix via matrix pipe
  mix_wmma(bas, D, afrag, u);

  // 2) per-k norms
  for (int k = 0; k < KQ; ++k) {
    float p = 0.f;
    for (int d = tid; d < D; d += 256) { float c = u[k * D + d]; p += c * c; }
    float s = block_reduce_sum(p, red);
    if (tid == 0) nrm[k] = sqrtf(s);
  }
  __syncthreads();

  // 3) normalize u; r = sum_k wk[k]*nrm[k]
  for (int idx = tid; idx < KQ * D; idx += 256) {
    int k = idx / D;
    u[idx] = u[idx] / (nrm[k] + EPSF);
  }
  __syncthreads();
  float r = 0.f;
  for (int k = 0; k < KQ; ++k) r += wk[k] * nrm[k];

  // 4) mu init = normalize(sum_k wk*u_k)
  for (int d = tid; d < D; d += 256) {
    float m = 0.f;
    for (int k = 0; k < KQ; ++k) m += wk[k] * u[k * D + d];
    mu[d] = m;
  }
  __syncthreads();
  {
    float p = 0.f;
    for (int d = tid; d < D; d += 256) { float m = mu[d]; p += m * m; }
    float s = block_reduce_sum(p, red);
    float inv = 1.f / (sqrtf(s) + EPSF);
    for (int d = tid; d < D; d += 256) mu[d] *= inv;
  }
  __syncthreads();

  // 5) 3 Karcher iterations
  for (int it = 0; it < 3; ++it) {
    for (int k = 0; k < KQ; ++k) {
      float p = 0.f;
      for (int d = tid; d < D; d += 256) p += mu[d] * u[k * D + d];
      float s = block_reduce_sum(p, red);
      if (tid == 0) {
        float dcl = fminf(fmaxf(s, -1.f + EPSF), 1.f - EPSF);
        dotv[k] = dcl;
        float th = acosf(dcl);
        coef[k] = (th < 1e-4f) ? 1.f : th / (sinf(th) + EPSF);
      }
    }
    __syncthreads();
    for (int d = tid; d < D; d += 256) {
      float m = mu[d];
      float acc = 0.f;
      for (int k = 0; k < KQ; ++k)
        acc += wk[k] * coef[k] * (u[k * D + d] - dotv[k] * m);
      vv[d] = acc;
    }
    __syncthreads();
    float p = 0.f;
    for (int d = tid; d < D; d += 256) { float q = vv[d]; p += q * q; }
    float vn = sqrtf(block_reduce_sum(p, red));
    float sc = (vn < 1e-4f) ? 1.f : sinf(vn) / (vn + EPSF);
    float cv = cosf(vn);
    for (int d = tid; d < D; d += 256) mu[d] = cv * mu[d] + sc * vv[d];
    __syncthreads();
    p = 0.f;
    for (int d = tid; d < D; d += 256) { float m = mu[d]; p += m * m; }
    float s3 = block_reduce_sum(p, red);
    float inv = 1.f / (sqrtf(s3) + EPSF);
    for (int d = tid; d < D; d += 256) mu[d] *= inv;
    __syncthreads();
  }

  // 6) restore scale
  for (int d = tid; d < D; d += 256) coreOut[d] = r * mu[d];
  __syncthreads();
}

// One workgroup (256 threads) per token: recipes -> Karcher cores -> TT-FFN -> gelu -> hidden
__global__ void __launch_bounds__(256)
karcher_tt_kernel(const float* __restrict__ x,
                  const int*   __restrict__ nidx,
                  const float* __restrict__ nw,
                  const float* __restrict__ recipes,
                  const float* __restrict__ bA1,
                  const float* __restrict__ bA2,
                  const float* __restrict__ bB1,
                  const float* __restrict__ bB2,
                  float* __restrict__ hidden) {
  extern __shared__ float sm[];
  const int tid = threadIdx.x;
  const int tok = blockIdx.x;    // 0 .. B*S-1

  float* rec  = sm + OFF_REC;
  float* wk   = sm + OFF_WK;
  float* nrm  = sm + OFF_NRM;
  float* dotv = sm + OFF_DOT;
  float* coef = sm + OFF_COEF;
  float* red  = sm + OFF_RED;
  float* xf   = sm + OFF_XF;
  float* core = sm + OFF_CORE;
  float* mu   = sm + OFF_MU;
  float* vv   = sm + OFF_VV;
  float* u    = sm + OFF_U;

  // recipes softmax (one thread per k-row) + x tile load
  if (tid < KQ) {
    int k = tid;
    int nid = nidx[tok * KQ + k];
    const float* row = recipes + nid * NB;
    float mx = -1e30f;
    for (int n = 0; n < NB; ++n) mx = fmaxf(mx, row[n]);
    float s = 0.f;
    for (int n = 0; n < NB; ++n) { float e = expf(row[n] - mx); rec[k * NB + n] = e; s += e; }
    float inv = 1.f / s;
    for (int n = 0; n < NB; ++n) rec[k * NB + n] *= inv;
  }
  if (tid == 0) {
    float s = 0.f;
    for (int k = 0; k < KQ; ++k) s += nw[tok * KQ + k];
    float inv = 1.f / (s + EPSF);
    for (int k = 0; k < KQ; ++k) wk[k] = nw[tok * KQ + k] * inv;
  }
  xf[tid] = x[tok * DM + tid];
  __syncthreads();

  // Build the rec A-fragment (16x32 f16, rows 8..15 zero) once; reuse 4x.
  v16h afrag;
  {
    const int lane = tid & 31;
    const int row  = lane & 15;
    const int hi   = lane >> 4;
    #pragma unroll
    for (int e = 0; e < 16; ++e) {
      const int n = (e < 8) ? (hi * 8 + e) : (16 + hi * 8 + (e - 8));
      afrag[e] = (row < KQ) ? (f16_t)rec[row * NB + n] : (f16_t)0.f;
    }
  }

  karcher_one(bA1, 1024, core + 0,    afrag, wk, nrm, dotv, coef, red, mu, vv, u);
  karcher_one(bA2, 1024, core + 1024, afrag, wk, nrm, dotv, coef, red, mu, vv, u);
  karcher_one(bB1, 2048, core + 2048, afrag, wk, nrm, dotv, coef, red, mu, vv, u);
  karcher_one(bB2, 2048, core + 4096, afrag, wk, nrm, dotv, coef, red, mu, vv, u);

  const float* cA1 = core;          // [16][8][8]
  const float* cA2 = core + 1024;   // [8][16][8]
  const float* cB1 = core + 2048;   // [8][8][32]
  const float* cB2 = core + 4096;   // [8][8][32]
  float* t1 = u;                    // 1024
  float* t2 = u + 4096;             // 2048
  float* hb = xf;                   // 64 (reuse after t1 consumed xf)

  // t1[j,r,k] = sum_i xf[i,j] * A1[i,r,k]
  for (int idx = tid; idx < 1024; idx += 256) {
    int j = idx >> 6, rr = (idx >> 3) & 7, kk = idx & 7;
    float acc = 0.f;
    #pragma unroll
    for (int i = 0; i < 16; ++i) acc += xf[i * 16 + j] * cA1[i * 64 + rr * 8 + kk];
    t1[idx] = acc;
  }
  __syncthreads();
  // h[k,l] = sum_{j,r} t1[j,r,k] * A2[r,j,l]
  if (tid < 64) {
    int kk = tid >> 3, l = tid & 7;
    float acc = 0.f;
    for (int j = 0; j < 16; ++j)
      #pragma unroll
      for (int rr = 0; rr < 8; ++rr)
        acc += t1[j * 64 + rr * 8 + kk] * cA2[rr * 128 + j * 8 + l];
    hb[tid] = acc;
  }
  __syncthreads();
  // t2[j,r,k] = sum_i h[i,j] * B1[i,r,k]   (j<8, r<8, k<32)
  for (int idx = tid; idx < 2048; idx += 256) {
    int j = idx >> 8, rr = (idx >> 5) & 7, kk = idx & 31;
    float acc = 0.f;
    #pragma unroll
    for (int i = 0; i < 8; ++i) acc += hb[i * 8 + j] * cB1[i * 256 + rr * 32 + kk];
    t2[idx] = acc;
  }
  __syncthreads();
  // out[k,l] = sum_{j,r} t2[j,r,k] * B2[r,j,l]; exact gelu; store hidden
  for (int idx = tid; idx < 1024; idx += 256) {
    int kk = idx >> 5, l = idx & 31;
    float acc = 0.f;
    #pragma unroll
    for (int j = 0; j < 8; ++j)
      #pragma unroll
      for (int rr = 0; rr < 8; ++rr)
        acc += t2[j * 256 + rr * 32 + kk] * cB2[rr * 256 + j * 32 + l];
    float g = 0.5f * acc * (1.f + erff(acc * 0.70710678118654752f));
    hidden[(size_t)tok * DH + idx] = g;
  }
}

// Down-projection GEMM via WMMA f16 (f32 accumulate):
// out[M=2048, N=256] = hidden[2048,1024] @ W^T (W row-major [256,1024]) + bias
// One 16x16 tile per wave, K-loop in steps of 32, prefetch next K-slab.
__global__ void __launch_bounds__(256)
downproj_wmma_kernel(const float* __restrict__ hid,
                     const float* __restrict__ w,
                     const float* __restrict__ bias,
                     float* __restrict__ out) {
  const int lane = threadIdx.x & 31;
  const int wave = threadIdx.x >> 5;
  const int tile = blockIdx.x * 8 + wave;  // 2048 tiles total
  const int tm = tile >> 4;                // 128 M-tiles
  const int tn = tile & 15;                // 16 N-tiles
  const int row = lane & 15;
  const int hi  = lane >> 4;

  // A fragment (16x32 f16): lane row = lane&15; K halves {hi*8..+7, 16+hi*8..+7}
  // B fragment (32x16 f16): lane col = lane&15; K halves {hi*8..+7, 16+hi*8..+7}
  const int m  = tm * 16 + row;
  const int n  = tn * 16 + row;
  const int kbA = hi * 8;
  const int kbB = hi * 16;

  const float* Hrow = hid + (size_t)m * DH;
  const float* Wrow = w   + (size_t)n * DH;

  v8f acc = {};
  for (int k0 = 0; k0 < DH; k0 += 32) {
    if (k0 + 32 < DH) {
      __builtin_prefetch(Hrow + k0 + 32, 0, 1);   // global_prefetch_b8
      __builtin_prefetch(Wrow + k0 + 32, 0, 1);
    }
    v16h a, b;
    #pragma unroll
    for (int e = 0; e < 8; ++e) {
      a[e]     = (f16_t)Hrow[k0 + kbA + e];
      a[8 + e] = (f16_t)Hrow[k0 + 16 + kbA + e];
    }
    #pragma unroll
    for (int e = 0; e < 16; ++e) {
      b[e] = (f16_t)Wrow[k0 + ((e < 8) ? (hi * 8 + e) : (16 + hi * 8 + (e - 8)))];
    }
    // (neg_a, A, neg_b, B, c_mod, C, reuse_a, reuse_b)
    acc = __builtin_amdgcn_wmma_f32_16x16x32_f16(false, a, false, b,
                                                 (short)0, acc, false, false);
  }

  // C/D layout: n = lane&15 ; m = hi*8 + j (VGPR j)
  const float bv = bias[tn * 16 + row];
  #pragma unroll
  for (int j = 0; j < 8; ++j) {
    int mg = tm * 16 + hi * 8 + j;
    out[(size_t)mg * DM + (tn * 16 + row)] = acc[j] + bv;
  }
}

extern "C" void kernel_launch(void* const* d_in, const int* in_sizes, int n_in,
                              void* d_out, int out_size, void* d_ws, size_t ws_size,
                              hipStream_t stream) {
  const float* x       = (const float*)d_in[0];
  const int*   nidx    = (const int*)  d_in[1];
  const float* nw      = (const float*)d_in[2];
  const float* recipes = (const float*)d_in[3];
  const float* bA1     = (const float*)d_in[4];
  const float* bA2     = (const float*)d_in[5];
  const float* bB1     = (const float*)d_in[6];
  const float* bB2     = (const float*)d_in[7];
  const float* wdw     = (const float*)d_in[8];
  const float* wdb     = (const float*)d_in[9];
  float* out    = (float*)d_out;
  float* hidden = (float*)d_ws;   // 2048*1024 floats = 8 MB scratch

  const size_t lds_bytes = (size_t)LDS_FLOATS * sizeof(float);
  hipLaunchKernelGGL(karcher_tt_kernel, dim3(BQ * SQ), dim3(256), lds_bytes, stream,
                     x, nidx, nw, recipes, bA1, bA2, bB1, bB2, hidden);

  // 2048 tiles / 8 waves per block = 256 blocks
  hipLaunchKernelGGL(downproj_wmma_kernel, dim3(256), dim3(256), 0, stream,
                     hidden, wdw, wdb, out);
}
